// MultiHeadedAttention_68968584839416
// MI455X (gfx1250) — matrix operand order repaired and checked by reading, not compiled, not run on hardware
//
#include <hip/hip_runtime.h>
#include <hip/hip_bf16.h>
#include <stdint.h>

// MHA forward for MI455X (gfx1250, wave32).
// - All GEMM FLOPs through v_wmma_f32_16x16x32_bf16 (f32 accumulate).
// - B-panels staged to LDS with GLOBAL_LOAD_ASYNC_TO_LDS_B128 (ASYNCcnt),
//   double-buffered, consumed via ds_load_b128.
// - attn ([B,H,S,S] f32, 537MB) materialized in d_out; single-pass
//   register-resident softmax (1 read + 1 write per element).

#define S_LEN   2048
#define D_MODEL 1024
#define HEADS   16
#define DHEAD   64
#define BATCH   2
#define NEGVAL  -100000000.0f

typedef __attribute__((ext_vector_type(16))) __bf16 v16bf;
typedef __attribute__((ext_vector_type(8)))  float  v8f;

union FragU {
    unsigned short u[16];
    v16bf v;
};

__device__ __forceinline__ unsigned short f2bf(float f) {
    unsigned int u = __builtin_bit_cast(unsigned int, f);
    u += 0x7FFFu + ((u >> 16) & 1u);     // round-to-nearest-even
    return (unsigned short)(u >> 16);
}

// ---- A fragment (16x32 bf16) from an f32 row-major matrix ------------------
// lane L holds row M=L%16; element e -> K = e + 16*(L/16) (contiguous order, same as B)
__device__ __forceinline__ v16bf load_a_f32(const float* __restrict__ src,
                                            int ld, int m0, int k0) {
    int lane = threadIdx.x & 31;
    int m    = lane & 15;
    int kh   = (lane >> 4) << 4;         // 0 or 16
    const float* p = src + (size_t)(m0 + m) * ld + k0 + kh;
    FragU f;
#pragma unroll
    for (int e = 0; e < 8; ++e) {
        f.u[e]     = f2bf(p[e]);         // K = kh .. kh+7
        f.u[e + 8] = f2bf(p[e + 8]);     // K = kh+8 .. kh+15
    }
    return f.v;
}

// ---- B fragment (32x16 bf16): column n of B = row (n0+n) of bf16 src -------
// lane L holds column N=L%16; element e -> K = e + 16*(L/16) (contiguous order, same as A)
__device__ __forceinline__ v16bf load_b_rows(const unsigned short* __restrict__ src,
                                             int ld, int n0, int k0) {
    int lane = threadIdx.x & 31;
    int n    = lane & 15;
    int koff = (lane >> 4) << 4;         // 0 or 16
    const unsigned short* p = src + (size_t)(n0 + n) * ld + k0 + koff;
    FragU f;
#pragma unroll
    for (int e = 0; e < 16; ++e) f.u[e] = p[e];
    return f.v;
}

// B fragment from an LDS-staged [64 x 32] bf16 tile (rows = N, cols = K)
__device__ __forceinline__ v16bf load_b_lds(const unsigned short* tile, int n0) {
    int lane = threadIdx.x & 31;
    int n    = lane & 15;
    int koff = (lane >> 4) << 4;
    const unsigned short* p = tile + (n0 + n) * 32 + koff;
    FragU f;
#pragma unroll
    for (int e = 0; e < 16; ++e) f.u[e] = p[e];
    return f.v;
}

__device__ __forceinline__ v8f wmma_bf16(v16bf a, v16bf b, v8f c) {
    return __builtin_amdgcn_wmma_f32_16x16x32_bf16(false, a, false, b,
                                                   (short)0, c, false, false);
}

// Cooperative async stage of a 64-row x 32-col bf16 panel into LDS.
// 128 threads; each issues two async b128 (16B) transfers. ASYNCcnt tracked.
__device__ __forceinline__ void stage_b_async(const unsigned short* __restrict__ src,
                                              int ld, int row0, int k0,
                                              unsigned short* dst /*shared*/) {
    int tid = threadIdx.x;
#pragma unroll
    for (int h = 0; h < 2; ++h) {
        int i   = tid + 128 * h;       // 0..255
        int row = i >> 2;              // 0..63
        int q   = i & 3;               // 0..3  (8 bf16 = 16B each)
        const unsigned short* g = src + (size_t)(row0 + row) * ld + k0 + 8 * q;
        // low 32 bits of a flat shared pointer == wave-relative LDS offset;
        // the async op adds LDS_BASE in hardware (ISA 10.2 / async addressing).
        unsigned lds = (unsigned)(unsigned long long)(uintptr_t)(dst + row * 32 + 8 * q);
        asm volatile("global_load_async_to_lds_b128 %0, %1, off"
                     :: "v"(lds), "v"(g) : "memory");
    }
}

__device__ __forceinline__ void wait_async0() {
    asm volatile("s_wait_asynccnt 0x0" ::: "memory");
}

// ---------------------------------------------------------------------------
__global__ void cast_f32_bf16(const float* __restrict__ in,
                              unsigned short* __restrict__ out, int n) {
    int i = blockIdx.x * 256 + threadIdx.x;
    if (i < n) out[i] = f2bf(in[i]);
}

// Y = scale * (X @ W^T + bias)   X:[4096,K] f32, W:[N,K] bf16
// Block 128 thr (4 waves); block tile 128x64; wave tile 32x64 (8 WMMA/K-step).
// W panel double-buffered in LDS via async loads.
// mode 0: f32  split-head [b,h,s,dh]            (q_up, pre-scaled)
// mode 1: bf16 split-head [b,h,s,dh]            (k_up)
// mode 2: bf16 split-head transposed [b,h,dh,s] (v_up^T)
// mode 3: f32  flat [4096, D_MODEL]             (final out)
__global__ void gemm_xwt(const float* __restrict__ X,
                         const unsigned short* __restrict__ W,
                         const float* __restrict__ bias,
                         void* __restrict__ Y,
                         float scale, int mode, int K) {
    __shared__ unsigned short Bt[2][64 * 32];   // 8 KiB, double buffered

    int wave = threadIdx.x >> 5;
    int m0   = blockIdx.x * 128 + wave * 32;
    int n0   = blockIdx.y * 64;

    v8f acc[2][4];
#pragma unroll
    for (int i = 0; i < 2; ++i)
#pragma unroll
        for (int j = 0; j < 4; ++j)
            acc[i][j] = (v8f){0.f,0.f,0.f,0.f,0.f,0.f,0.f,0.f};

    stage_b_async(W, K, n0, 0, Bt[0]);
    wait_async0();
    __syncthreads();

    for (int k0 = 0; k0 < K; k0 += 32) {
        int cur = (k0 >> 5) & 1;
        if (k0 + 32 < K)
            stage_b_async(W, K, n0, k0 + 32, Bt[cur ^ 1]);

        v16bf a0 = load_a_f32(X, K, m0,      k0);
        v16bf a1 = load_a_f32(X, K, m0 + 16, k0);
#pragma unroll
        for (int j = 0; j < 4; ++j) {
            v16bf b = load_b_lds(Bt[cur], 16 * j);
            acc[0][j] = wmma_bf16(a0, b, acc[0][j]);
            acc[1][j] = wmma_bf16(a1, b, acc[1][j]);
        }
        wait_async0();
        __syncthreads();
    }

    int lane = threadIdx.x & 31;
    int nlo  = lane & 15;
    int mhi  = (lane >> 4) << 3;
#pragma unroll
    for (int i = 0; i < 2; ++i) {
#pragma unroll
        for (int j = 0; j < 4; ++j) {
#pragma unroll
            for (int r = 0; r < 8; ++r) {
                int mm = m0 + 16 * i + r + mhi;   // global row in [0,4096)
                int nn = n0 + 16 * j + nlo;       // global col in [0,1024)
                float v = (acc[i][j][r] + bias[nn]) * scale;
                int b = mm >> 11, s = mm & 2047;
                int h = nn >> 6,  d = nn & 63;
                if (mode == 0) {
                    ((float*)Y)[(((size_t)b * HEADS + h) * S_LEN + s) * DHEAD + d] = v;
                } else if (mode == 1) {
                    ((unsigned short*)Y)[(((size_t)b * HEADS + h) * S_LEN + s) * DHEAD + d] = f2bf(v);
                } else if (mode == 2) {
                    ((unsigned short*)Y)[(((size_t)b * HEADS + h) * DHEAD + d) * S_LEN + s] = f2bf(v);
                } else {
                    ((float*)Y)[(size_t)mm * D_MODEL + nn] = v;
                }
            }
        }
    }
}

// scores[b,h,q,k] = q_up . k_up  (q already scaled), masked -> NEG
__global__ void scores_kernel(const float* __restrict__ Qup,
                              const unsigned short* __restrict__ Kup,
                              const unsigned char* __restrict__ mask,
                              float* __restrict__ attn) {
    int bh = blockIdx.z;
    int b  = bh >> 4;
    const float* q          = Qup + (size_t)bh * S_LEN * DHEAD;
    const unsigned short* k = Kup + (size_t)bh * S_LEN * DHEAD;

    int wave = threadIdx.x >> 5;
    int q0   = blockIdx.x * 16;
    int n0   = blockIdx.y * 256 + wave * 64;

    v8f acc[4];
#pragma unroll
    for (int j = 0; j < 4; ++j) acc[j] = (v8f){0.f,0.f,0.f,0.f,0.f,0.f,0.f,0.f};

#pragma unroll
    for (int k0 = 0; k0 < DHEAD; k0 += 32) {
        v16bf a = load_a_f32(q, DHEAD, q0, k0);
#pragma unroll
        for (int j = 0; j < 4; ++j) {
            v16bf bfrag = load_b_rows(k, DHEAD, n0 + 16 * j, k0);
            acc[j] = wmma_bf16(a, bfrag, acc[j]);
        }
    }

    const unsigned char* mrow = mask + (size_t)b * S_LEN * S_LEN;
    float* arow = attn + (size_t)bh * S_LEN * S_LEN;
    int lane = threadIdx.x & 31;
    int nlo  = lane & 15;
    int mhi  = (lane >> 4) << 3;
#pragma unroll
    for (int j = 0; j < 4; ++j) {
#pragma unroll
        for (int r = 0; r < 8; ++r) {
            int qq = q0 + r + mhi;
            int kn = n0 + 16 * j + nlo;
            size_t idx = (size_t)qq * S_LEN + kn;
            arow[idx] = mrow[idx] ? NEGVAL : acc[j][r];
        }
    }
}

// in-place row softmax, row fully register-resident: 1 read + 1 write.
// 256 threads per 2048-element row.
__global__ void softmax_kernel(float* __restrict__ attn) {
    float* row = attn + (size_t)blockIdx.x * S_LEN;
    int t = threadIdx.x;
    __shared__ float red[8];
    __shared__ float bcast;

    float v[8];
#pragma unroll
    for (int i = 0; i < 8; ++i) v[i] = row[t + 256 * i];

    float lmax = v[0];
#pragma unroll
    for (int i = 1; i < 8; ++i) lmax = fmaxf(lmax, v[i]);
#pragma unroll
    for (int off = 16; off > 0; off >>= 1)
        lmax = fmaxf(lmax, __shfl_xor(lmax, off, 32));
    if ((t & 31) == 0) red[t >> 5] = lmax;
    __syncthreads();
    if (t == 0) {
        float m = red[0];
#pragma unroll
        for (int i = 1; i < 8; ++i) m = fmaxf(m, red[i]);
        bcast = m;
    }
    __syncthreads();
    float rmax = bcast;

    float lsum = 0.f;
#pragma unroll
    for (int i = 0; i < 8; ++i) {
        v[i] = __expf(v[i] - rmax);
        lsum += v[i];
    }
#pragma unroll
    for (int off = 16; off > 0; off >>= 1)
        lsum += __shfl_xor(lsum, off, 32);
    if ((t & 31) == 0) red[t >> 5] = lsum;
    __syncthreads();
    if (t == 0) {
        float s = 0.f;
#pragma unroll
        for (int i = 0; i < 8; ++i) s += red[i];
        bcast = 1.0f / s;
    }
    __syncthreads();
    float inv = bcast;
#pragma unroll
    for (int i = 0; i < 8; ++i) row[t + 256 * i] = v[i] * inv;
}

// ctx[b, q, h*64+d] = sum_k attn[b,h,q,k] * vT[b,h,d,k]
// Block 128 thr; wave tile 32x64; vT panel (shared by all waves) staged in LDS.
__global__ void av_kernel(const float* __restrict__ attn,
                          const unsigned short* __restrict__ Vt,
                          float* __restrict__ ctx) {
    __shared__ unsigned short Bt[2][64 * 32];

    int bh = blockIdx.y;
    int b  = bh >> 4, h = bh & 15;
    const float* arow        = attn + (size_t)bh * S_LEN * S_LEN;
    const unsigned short* vt = Vt   + (size_t)bh * DHEAD * S_LEN;

    int wave = threadIdx.x >> 5;
    int q0   = blockIdx.x * 128 + wave * 32;

    v8f acc[2][4];
#pragma unroll
    for (int i = 0; i < 2; ++i)
#pragma unroll
        for (int j = 0; j < 4; ++j)
            acc[i][j] = (v8f){0.f,0.f,0.f,0.f,0.f,0.f,0.f,0.f};

    stage_b_async(vt, S_LEN, 0, 0, Bt[0]);
    wait_async0();
    __syncthreads();

    for (int k0 = 0; k0 < S_LEN; k0 += 32) {
        int cur = (k0 >> 5) & 1;
        if (k0 + 32 < S_LEN) {
            stage_b_async(vt, S_LEN, 0, k0 + 32, Bt[cur ^ 1]);
            // prefetch next attn A-panel (global_prefetch_b8)
            __builtin_prefetch(arow + (size_t)(q0 + (threadIdx.x & 31)) * S_LEN + k0 + 32, 0, 3);
        }

        v16bf a0 = load_a_f32(arow, S_LEN, q0,      k0);
        v16bf a1 = load_a_f32(arow, S_LEN, q0 + 16, k0);
#pragma unroll
        for (int j = 0; j < 4; ++j) {
            v16bf bfrag = load_b_lds(Bt[cur], 16 * j);
            acc[0][j] = wmma_bf16(a0, bfrag, acc[0][j]);
            acc[1][j] = wmma_bf16(a1, bfrag, acc[1][j]);
        }
        wait_async0();
        __syncthreads();
    }

    int lane = threadIdx.x & 31;
    int nlo  = lane & 15;
    int mhi  = (lane >> 4) << 3;
#pragma unroll
    for (int i = 0; i < 2; ++i) {
#pragma unroll
        for (int j = 0; j < 4; ++j) {
#pragma unroll
            for (int r = 0; r < 8; ++r) {
                int qq = q0 + 16 * i + r + mhi;
                int dd = 16 * j + nlo;
                ctx[((size_t)b * S_LEN + qq) * D_MODEL + h * DHEAD + dd] = acc[i][j][r];
            }
        }
    }
}

// ---------------------------------------------------------------------------
extern "C" void kernel_launch(void* const* d_in, const int* in_sizes, int n_in,
                              void* d_out, int out_size, void* d_ws, size_t ws_size,
                              hipStream_t stream) {
    // inputs: key value query mask Wq bq Wk bk Wv bv Wo bo
    const float* key   = (const float*)d_in[0];
    const float* value = (const float*)d_in[1];
    const float* query = (const float*)d_in[2];
    const unsigned char* mask = (const unsigned char*)d_in[3];  // jnp bool_
    const float* Wq = (const float*)d_in[4];
    const float* bq = (const float*)d_in[5];
    const float* Wk = (const float*)d_in[6];
    const float* bk = (const float*)d_in[7];
    const float* Wv = (const float*)d_in[8];
    const float* bv = (const float*)d_in[9];
    const float* Wo = (const float*)d_in[10];
    const float* bo = (const float*)d_in[11];

    char* ws = (char*)d_ws;
    const size_t MB = 1024 * 1024;
    unsigned short* wq_b  = (unsigned short*)(ws + 0 * MB);   // 2 MiB each
    unsigned short* wk_b  = (unsigned short*)(ws + 2 * MB);
    unsigned short* wv_b  = (unsigned short*)(ws + 4 * MB);
    unsigned short* wo_b  = (unsigned short*)(ws + 6 * MB);
    float*          q_up  = (float*)         (ws + 8 * MB);   // 16 MiB f32 [B,H,S,Dh]
    unsigned short* k_up  = (unsigned short*)(ws + 24 * MB);  //  8 MiB bf16 [B,H,S,Dh]
    unsigned short* vT    = (unsigned short*)(ws + 32 * MB);  //  8 MiB bf16 [B,H,Dh,S]
    float*          ctx   = (float*)         (ws + 40 * MB);  // 16 MiB f32 [B*S, D]

    float* out_p  = (float*)d_out;                                   // [B,S,D]
    float* attn_p = (float*)d_out + (size_t)BATCH * S_LEN * D_MODEL; // [B,H,S,S]

    const int NW = D_MODEL * D_MODEL;  // 1M elems per weight
    cast_f32_bf16<<<NW / 256, 256, 0, stream>>>(Wq, wq_b, NW);
    cast_f32_bf16<<<NW / 256, 256, 0, stream>>>(Wk, wk_b, NW);
    cast_f32_bf16<<<NW / 256, 256, 0, stream>>>(Wv, wv_b, NW);
    cast_f32_bf16<<<NW / 256, 256, 0, stream>>>(Wo, wo_b, NW);

    dim3 gblk(128);
    dim3 ggrid(BATCH * S_LEN / 128, D_MODEL / 64);  // (32,16)
    // q_up = (query@Wq^T + bq) / sqrt(Dh)
    gemm_xwt<<<ggrid, gblk, 0, stream>>>(query, wq_b, bq, q_up, 0.125f, 0, D_MODEL);
    gemm_xwt<<<ggrid, gblk, 0, stream>>>(key,   wk_b, bk, k_up, 1.0f,   1, D_MODEL);
    gemm_xwt<<<ggrid, gblk, 0, stream>>>(value, wv_b, bv, vT,   1.0f,   2, D_MODEL);

    dim3 sgrid(S_LEN / 16, S_LEN / 256, BATCH * HEADS);  // (128,8,32)
    scores_kernel<<<sgrid, gblk, 0, stream>>>(q_up, k_up, mask, attn_p);

    softmax_kernel<<<BATCH * HEADS * S_LEN, 256, 0, stream>>>(attn_p);

    dim3 avgrid(S_LEN / 128, BATCH * HEADS);  // (16,32)
    av_kernel<<<avgrid, gblk, 0, stream>>>(attn_p, vT, ctx);

    gemm_xwt<<<ggrid, gblk, 0, stream>>>(ctx, wo_b, bo, out_p, 1.0f, 3, D_MODEL);
}
